// GAT_14766097563858
// MI455X (gfx1250) — matrix operand order repaired and checked by reading, not compiled, not run on hardware
//
#include <hip/hip_runtime.h>

// ---------------- problem constants ----------------
#define N_NODES 50000
#define E_RAW   400000
#define E_TOT   450000          // + self loops
#define FOUT    256             // HEADS*HID
#define OUTD    32
#define NEGSLOPE 0.2f

typedef __attribute__((ext_vector_type(16))) __bf16 v16bf;
typedef __attribute__((ext_vector_type(8)))  float  v8f;

// ---------------- helpers ----------------
__device__ inline unsigned short f2bfbits(float f) {   // RNE f32->bf16 bits
  unsigned u = __float_as_uint(f);
  return (unsigned short)((u + 0x7fffu + ((u >> 16) & 1u)) >> 16);
}
// order-preserving f32 <-> u32 map for deterministic atomicMax
__device__ inline unsigned mapf(float f) {
  unsigned u = __float_as_uint(f);
  return (u & 0x80000000u) ? ~u : (u | 0x80000000u);
}
__device__ inline float unmapf(unsigned u) {
  return __uint_as_float((u & 0x80000000u) ? (u & 0x7fffffffu) : ~u);
}
__device__ inline float wsum(float v) {
#pragma unroll
  for (int o = 16; o; o >>= 1) v += __shfl_xor(v, o, 32);
  return v;
}

// ---------------- generic zero ----------------
__global__ void k_zero(float* p, long n) {
  long i = (long)blockIdx.x * blockDim.x + threadIdx.x;
  if (i < n) p[i] = 0.0f;
}

// ---------------- f32 -> bf16 pack (one-time per layer) ----------------
__global__ void k_pack(const float* __restrict__ src, unsigned short* __restrict__ dst,
                       long n) {
  long i = (long)blockIdx.x * blockDim.x + threadIdx.x;
  if (i < n) dst[i] = f2bfbits(src[i]);
}

// ---------------- GEMM: Y[N,256] = Xb[N,FIN] @ Wb[FIN,256]  (bf16 WMMA, f32 acc) --
// One wave per 16x16 output tile. A/B pre-packed bf16 -> inner loop is
// 2x global_load_b128 (A) + 16 u16 loads (B) + 1x v_wmma_f32_16x16x32_bf16, fully unrolled over K.
template <int FIN>
__global__ __launch_bounds__(256) void k_gemm(const unsigned short* __restrict__ Xb,
                                              const unsigned short* __restrict__ Wb,
                                              float* __restrict__ Y) {
  int wid  = (blockIdx.x * blockDim.x + threadIdx.x) >> 5;   // 0..49999
  int lane = threadIdx.x & 31;
  int mtile = wid >> 4;            // 0..3124
  int ntile = wid & 15;            // 0..15
  int half  = lane >> 4;
  const unsigned short* xrow = Xb + (size_t)(mtile * 16 + (lane & 15)) * FIN;
  const int ncol = ntile * 16;
  v8f c = {0.f, 0.f, 0.f, 0.f, 0.f, 0.f, 0.f, 0.f};
#pragma unroll
  for (int k0 = 0; k0 < FIN; k0 += 32) {
    union { v16bf v; uint4 q[2]; unsigned short u[16]; } A, B;
    // A 16x32 bf16 layout: half 0 -> K {0..7}U{16..23}; half 1 -> {8..15}U{24..31}
    A.q[0] = *(const uint4*)(xrow + k0 + half * 8);
    A.q[1] = *(const uint4*)(xrow + k0 + 16 + half * 8);
    // B 32x16: lane = N col (lane&15); same K layout as A
#pragma unroll
    for (int s = 0; s < 16; ++s)
      B.u[s] = Wb[(size_t)(k0 + (s < 8 ? half * 8 + s : 16 + half * 8 + s - 8)) * FOUT + ncol + (lane & 15)];
    c = __builtin_amdgcn_wmma_f32_16x16x32_bf16(false, A.v, false, B.v,
                                                (short)0, c, false, false);
  }
  // C/D: vgpr r -> row = 16*mtile + 8*half + r ; col = ncol + (lane&15)
  int row0 = mtile * 16 + half * 8;
  int col  = ncol + (lane & 15);
#pragma unroll
  for (int r = 0; r < 8; ++r) Y[(size_t)(row0 + r) * FOUT + col] = c[r];
}

// ---------------- attention logits: al_{src,dst}[n][h] = sum_c xW[n,h,c]*a[h,c] ---
__global__ __launch_bounds__(256) void k_al(const float* __restrict__ XW,
                                            const float* __restrict__ a_src,
                                            const float* __restrict__ a_dst,
                                            float* __restrict__ alsrc,
                                            float* __restrict__ aldst) {
  int wid  = (blockIdx.x * blockDim.x + threadIdx.x) >> 5;   // node id
  int lane = threadIdx.x & 31;
  const float* row = XW + (size_t)wid * FOUT;
  float as[4] = {0.f, 0.f, 0.f, 0.f};
  float ad[4] = {0.f, 0.f, 0.f, 0.f};
#pragma unroll
  for (int j = 0; j < 8; ++j) {
    int d = lane + 32 * j;
    int h = j >> 1;              // (lane+32j)>>6 is lane-independent
    float v = row[d];
    as[h] += v * a_src[d];
    ad[h] += v * a_dst[d];
  }
#pragma unroll
  for (int h = 0; h < 4; ++h) { as[h] = wsum(as[h]); ad[h] = wsum(ad[h]); }
  if (lane == 0) {
#pragma unroll
    for (int h = 0; h < 4; ++h) {
      alsrc[wid * 4 + h] = as[h];
      aldst[wid * 4 + h] = ad[h];
    }
  }
}

// ---------------- edge pass 1: segment max over dst ----------------
__global__ void k_emax(const int* __restrict__ ei, const float* __restrict__ alsrc,
                       const float* __restrict__ aldst, unsigned* __restrict__ mU) {
  int e = blockIdx.x * blockDim.x + threadIdx.x;
  if (e >= E_TOT) return;
  int s, d;
  if (e < E_RAW) { s = ei[e]; d = ei[E_RAW + e]; } else { s = d = e - E_RAW; }
#pragma unroll
  for (int h = 0; h < 4; ++h) {
    float v = alsrc[s * 4 + h] + aldst[d * 4 + h];
    v = (v >= 0.f) ? v : NEGSLOPE * v;
    atomicMax(&mU[d * 4 + h], mapf(v));
  }
}

// ---------------- edge pass 2: exp + segment sum ----------------
__global__ void k_esum(const int* __restrict__ ei, const float* __restrict__ alsrc,
                       const float* __restrict__ aldst, const unsigned* __restrict__ mU,
                       float* __restrict__ ssum, float* __restrict__ wbuf) {
  int e = blockIdx.x * blockDim.x + threadIdx.x;
  if (e >= E_TOT) return;
  int s, d;
  if (e < E_RAW) { s = ei[e]; d = ei[E_RAW + e]; } else { s = d = e - E_RAW; }
#pragma unroll
  for (int h = 0; h < 4; ++h) {
    float v = alsrc[s * 4 + h] + aldst[d * 4 + h];
    v = (v >= 0.f) ? v : NEGSLOPE * v;
    float w = expf(v - unmapf(mU[d * 4 + h]));
    wbuf[(size_t)e * 4 + h] = w;
    atomicAdd(&ssum[d * 4 + h], w);
  }
}

// ---------------- edge pass 3: out[dst] += alpha * xW[src] (warp per edge) -------
__global__ __launch_bounds__(256) void k_eagg(const int* __restrict__ ei,
                                              const float* __restrict__ XW,
                                              const float* __restrict__ wbuf,
                                              const float* __restrict__ ssum,
                                              float* __restrict__ OUT) {
  int e    = (blockIdx.x * blockDim.x + threadIdx.x) >> 5;   // edge id
  int lane = threadIdx.x & 31;
  int s, d;
  if (e < E_RAW) { s = ei[e]; d = ei[E_RAW + e]; } else { s = d = e - E_RAW; }
  float alpha[4];
#pragma unroll
  for (int h = 0; h < 4; ++h)
    alpha[h] = wbuf[(size_t)e * 4 + h] / ssum[d * 4 + h];
  const float* src = XW + (size_t)s * FOUT;
  float* dst = OUT + (size_t)d * FOUT;
#pragma unroll
  for (int j = 0; j < 8; ++j) {
    int c = lane + 32 * j;
    atomicAdd(&dst[c], src[c] * alpha[j >> 1]);
  }
}

// ---------------- bias + optional relu ----------------
__global__ void k_bias(float* __restrict__ OUT, const float* __restrict__ b, int relu) {
  long i = (long)blockIdx.x * blockDim.x + threadIdx.x;
  if (i >= (long)N_NODES * FOUT) return;
  float v = OUT[i] + b[i & (FOUT - 1)];
  OUT[i] = relu ? fmaxf(v, 0.f) : v;
}

// ---------------- pooling: gate = sigmoid(x.Wg+bg), block-reduced max -----------
__global__ __launch_bounds__(256) void k_gate(const float* __restrict__ X,
                                              const float* __restrict__ Wg,
                                              const float* __restrict__ bg,
                                              float* __restrict__ gate,
                                              unsigned* __restrict__ gmaxU, int fin) {
  int wid  = (blockIdx.x * blockDim.x + threadIdx.x) >> 5;   // node
  int lane = threadIdx.x & 31;
  const float* row = X + (size_t)wid * fin;
  float acc = 0.f;
  for (int f = lane; f < fin; f += 32) acc += row[f] * Wg[f];
  acc = wsum(acc);
  __shared__ float sm[8];
  if (lane == 0) {
    float g = 1.f / (1.f + expf(-(acc + bg[0])));
    gate[wid] = g;
    sm[threadIdx.x >> 5] = g;
  }
  __syncthreads();
  if (threadIdx.x == 0) {
    float m = sm[0];
#pragma unroll
    for (int w = 1; w < 8; ++w) m = fmaxf(m, sm[w]);
    atomicMax(gmaxU, mapf(m));
  }
}

__global__ void k_expsum(float* __restrict__ gate, const unsigned* __restrict__ gmaxU,
                         float* __restrict__ gsum) {
  int i = blockIdx.x * blockDim.x + threadIdx.x;
  float m = unmapf(*gmaxU);
  float e = 0.f;
  if (i < N_NODES) { e = expf(gate[i] - m); gate[i] = e; }
  float v = wsum(e);
  __shared__ float sm[8];
  if ((threadIdx.x & 31) == 0) sm[threadIdx.x >> 5] = v;
  __syncthreads();
  if (threadIdx.x == 0) {
    float t = 0.f;
#pragma unroll
    for (int w = 0; w < 8; ++w) t += sm[w];
    atomicAdd(gsum, t);
  }
}

// pooled[f] += sum_n e[n]*x[n,f]  (block handles 500 nodes, coalesced over f)
__global__ __launch_bounds__(256) void k_pool(const float* __restrict__ X,
                                              const float* __restrict__ gate,
                                              float* __restrict__ pooled, int fin) {
  int f   = threadIdx.x % fin;
  int sub = threadIdx.x / fin;
  int nsub = 256 / fin;
  int start = blockIdx.x * 500;
  float acc = 0.f;
  for (int n = start + sub; n < start + 500; n += nsub)
    acc += gate[n] * X[(size_t)n * fin + f];
  atomicAdd(&pooled[f], acc);
}

// hvec[k] += hw[l] * ((pooled/gsum) @ Wp + bp)[k]
__global__ void k_head(const float* __restrict__ pooled, const float* __restrict__ gsum,
                       const float* __restrict__ Wp, const float* __restrict__ bp,
                       const float* __restrict__ hw, int l, float* __restrict__ hvec,
                       int fin) {
  int k = threadIdx.x;                  // 32 threads
  float inv = 1.f / gsum[0];
  float acc = 0.f;
  for (int f = 0; f < fin; ++f) acc += pooled[f] * Wp[f * OUTD + k];
  hvec[k] += hw[l] * (acc * inv + bp[k]);
}

__global__ void k_final(const float* __restrict__ hvec, const float* __restrict__ beta,
                        const float* __restrict__ h0, float* __restrict__ out) {
  float v = hvec[threadIdx.x] * beta[threadIdx.x];
  v = wsum(v);
  if (threadIdx.x == 0) out[0] = v + h0[0];
}

// ---------------- launch ----------------
extern "C" void kernel_launch(void* const* d_in, const int* in_sizes, int n_in,
                              void* d_out, int out_size, void* d_ws, size_t ws_size,
                              hipStream_t stream) {
  const float* x   = (const float*)d_in[0];
  const int*   ei  = (const int*)d_in[1];
  // d_in[2] = batch (unused, single graph)
  // conv_params: l -> W=3+4l, a_src=4+4l, a_dst=5+4l, b=6+4l
  // gate_params: l -> Wg=15+2l, bg=16+2l
  // pred_params: l -> Wp=23+2l, bp=24+2l
  const float* hw   = (const float*)d_in[31];
  const float* h0   = (const float*)d_in[32];
  const float* beta = (const float*)d_in[33];

  const long NB = (long)N_NODES * FOUT;              // 12.8M elements
  // bf16 regions first (16B-aligned b128 loads)
  unsigned short* Xbf = (unsigned short*)d_ws;       // N*256 bf16 (25.6 MB)
  unsigned short* Wbf = Xbf + NB;                    // 256*256 bf16
  float* B0     = (float*)(Wbf + FOUT * FOUT);       // xW (f32)
  float* B1     = B0 + NB;                           // x / aggregated out
  float* alsrc  = B1 + NB;                           // N*4
  float* aldst  = alsrc + 4L * N_NODES;              // N*4
  float* mS     = aldst + 4L * N_NODES;              // N*4 (mapped u32 max)
  float* ssum   = mS    + 4L * N_NODES;              // N*4
  float* wbuf   = ssum  + 4L * N_NODES;              // E_TOT*4
  float* gate   = wbuf  + 4L * E_TOT;                // N
  float* pooled = gate  + N_NODES;                   // 256
  float* gsum   = pooled + FOUT;                     // 1
  float* gmax   = gsum + 1;                          // 1 (unsigned)
  float* hvec   = gmax + 1;                          // 32

  k_zero<<<1, 32, 0, stream>>>(hvec, 32);
  // pack layer-0 input x (N x 128) to bf16
  k_pack<<<25000, 256, 0, stream>>>(x, Xbf, (long)N_NODES * 128);

  for (int l = 0; l < 3; ++l) {
    const float* xc  = (l == 0) ? x : B1;
    int fin          = (l == 0) ? 128 : 256;
    const float* W   = (const float*)d_in[3 + 4 * l];
    const float* a_s = (const float*)d_in[4 + 4 * l];
    const float* a_d = (const float*)d_in[5 + 4 * l];
    const float* bcv = (const float*)d_in[6 + 4 * l];
    const float* Wg  = (const float*)d_in[15 + 2 * l];
    const float* bg  = (const float*)d_in[16 + 2 * l];
    const float* Wp  = (const float*)d_in[23 + 2 * l];
    const float* bp  = (const float*)d_in[24 + 2 * l];

    // ---- attention pool on current x (f32) ----
    k_zero  <<<2, 256, 0, stream>>>(pooled, 258);                 // pooled+gsum+gmax
    k_gate  <<<6250, 256, 0, stream>>>(xc, Wg, bg, gate, (unsigned*)gmax, fin);
    k_expsum<<<196, 256, 0, stream>>>(gate, (const unsigned*)gmax, gsum);
    k_pool  <<<100, 256, 0, stream>>>(xc, gate, pooled, fin);
    k_head  <<<1, 32, 0, stream>>>(pooled, gsum, Wp, bp, hw, l, hvec, fin);

    // ---- GAT conv ----
    k_pack<<<(int)(((long)fin * FOUT + 255) / 256), 256, 0, stream>>>(W, Wbf,
                                                                     (long)fin * FOUT);
    if (fin == 128) k_gemm<128><<<6250, 256, 0, stream>>>(Xbf, Wbf, B0);
    else            k_gemm<256><<<6250, 256, 0, stream>>>(Xbf, Wbf, B0);
    k_al  <<<6250, 256, 0, stream>>>(B0, a_s, a_d, alsrc, aldst);
    k_zero<<<(int)((NB + 255) / 256), 256, 0, stream>>>(B1, NB);  // xc(B1) dead post-GEMM
    k_zero<<<1563, 256, 0, stream>>>(mS, 8L * N_NODES);           // m (sentinel 0) + ssum
    k_emax<<<1758, 256, 0, stream>>>(ei, alsrc, aldst, (unsigned*)mS);
    k_esum<<<1758, 256, 0, stream>>>(ei, alsrc, aldst, (const unsigned*)mS, ssum, wbuf);
    k_eagg<<<56250, 256, 0, stream>>>(ei, B0, wbuf, ssum, B1);
    k_bias<<<50000, 256, 0, stream>>>(B1, bcv, (l < 2) ? 1 : 0);
    // pack next layer's x (B1, N x 256) to bf16 (also used only if l<2, harmless at l=2)
    if (l < 2) k_pack<<<50000, 256, 0, stream>>>(B1, Xbf, NB);
  }

  // ---- final pool (layer 3 params, fin=256) ----
  {
    const float* Wg = (const float*)d_in[15 + 6];
    const float* bg = (const float*)d_in[16 + 6];
    const float* Wp = (const float*)d_in[23 + 6];
    const float* bp = (const float*)d_in[24 + 6];
    k_zero  <<<2, 256, 0, stream>>>(pooled, 258);
    k_gate  <<<6250, 256, 0, stream>>>(B1, Wg, bg, gate, (unsigned*)gmax, 256);
    k_expsum<<<196, 256, 0, stream>>>(gate, (const unsigned*)gmax, gsum);
    k_pool  <<<100, 256, 0, stream>>>(B1, gate, pooled, 256);
    k_head  <<<1, 32, 0, stream>>>(pooled, gsum, Wp, bp, hw, 3, hvec, 256);
  }
  k_final<<<1, 32, 0, stream>>>(hvec, beta, h0, (float*)d_out);
}
